// DeltaRuleUpdate_23244363006471
// MI455X (gfx1250) — matrix operand order repaired and checked by reading, not compiled, not run on hardware
//
#include <hip/hip_runtime.h>

// Problem constants (from the reference)
#define Bc   4
#define Hc   16
#define Sc   8192
#define DKc  128
#define DVc  128
#define BHc  (Bc * Hc)

// Work decomposition
#define SPLIT  8                  // S-splits per (b,h)
#define CHUNK  (Sc / SPLIT)       // 1024 rows per workgroup
#define ST     32                 // s-tile = WMMA K dim
#define ITERS  (CHUNK / ST)       // 32 iterations

typedef __attribute__((ext_vector_type(16))) __bf16 v16bf;
typedef __attribute__((ext_vector_type(2)))  __bf16 v2bf;
typedef __attribute__((ext_vector_type(8)))  float  v8f;

union Frag { v16bf v; uint4 q[2]; };

// Native bf16 conversion: lowers to v_cvt_pk_bf16_f32-class ops on gfx1250
__device__ __forceinline__ unsigned pk2bf(float lo, float hi) {
  v2bf t; t.x = (__bf16)lo; t.y = (__bf16)hi;
  return __builtin_bit_cast(unsigned, t);
}
__device__ __forceinline__ unsigned short f2bf(float x) {
  return __builtin_bit_cast(unsigned short, (__bf16)x);
}
__device__ __forceinline__ float phi_elu1(float x) {
  // elu(x)+1 = x>0 ? x+1 : exp(x)
  return x > 0.f ? x + 1.f : __expf(x);
}

__global__ __launch_bounds__(256) void delta_rule_kernel(
    const float* __restrict__ K, const float* __restrict__ V,
    const float* __restrict__ M, float* __restrict__ outM,
    float* __restrict__ outZ)
{
  // LDS: 32KB M^T + 8KB phi + 8KB phi^T + 8KB E^T + 512B z  = ~56.5KB
  __shared__ __align__(16) unsigned short sMT  [DVc * DKc];  // M^T  bf16, [v][k]
  __shared__ __align__(16) unsigned short sPhi [ST  * DKc];  // phi  bf16, [s][k]
  __shared__ __align__(16) unsigned short sPhiT[DKc * ST ];  // phi^T bf16, [k][s]
  __shared__ __align__(16) unsigned short sET  [DVc * ST ];  // E^T  bf16, [v][s]
  __shared__ float sZ[DKc];

  const int tid  = threadIdx.x;
  const int wave = tid >> 5;          // 0..7
  const int lane = tid & 31;
  const int n    = lane & 15;         // B/C/D column within 16x16 tile
  const int hi   = lane >> 4;         // lane-half selector

  const int  bh    = blockIdx.x / SPLIT;
  const int  chunk = blockIdx.x % SPLIT;
  const long sbase = (long)chunk * CHUNK;

  const float* Kbh = K + (long)bh * Sc * DKc;
  const float* Vbh = V + (long)bh * Sc * DVc;
  const float* Mbh = M + (long)bh * DKc * DVc;

  // ---- Stage M^T (bf16) into LDS: coalesced global reads, one-time ----
  for (int idx = tid; idx < DKc * DVc; idx += 256) {
    int k = idx >> 7, v = idx & 127;
    sMT[v * DKc + k] = f2bf(Mbh[idx]);
  }
  if (tid < DKc) sZ[tid] = 0.f;

  // GEMM2 accumulators: wave w owns delta k-tile w (rows 16w..16w+15) x 8 v-tiles
  v8f accD[8];
#pragma unroll
  for (int i = 0; i < 8; ++i)
    accD[i] = (v8f){0.f, 0.f, 0.f, 0.f, 0.f, 0.f, 0.f, 0.f};

  // phi producer mapping: thread -> 4x4 sub-block (rows r0..r0+3, cols c0..c0+3)
  // so both phi layouts are written as contiguous 8B ds_store_b64's.
  const int r0 = (tid >> 5) * 4;      // s rows (8 blocks of 4 = 32)
  const int c0 = (tid & 31) * 4;      // k cols (32 blocks of 4 = 128)
  float zreg[4] = {0.f, 0.f, 0.f, 0.f};

  __syncthreads();

  for (int it = 0; it < ITERS; ++it) {
    const long srow0 = sbase + (long)it * ST;

    // ---- Phase 1: phi = elu(K)+1, dual-layout store, z accumulate ----
    {
      float p[4][4];
#pragma unroll
      for (int j = 0; j < 4; ++j) {
        float4 x = *(const float4*)(Kbh + (srow0 + r0 + j) * DKc + c0);
        p[j][0] = phi_elu1(x.x); p[j][1] = phi_elu1(x.y);
        p[j][2] = phi_elu1(x.z); p[j][3] = phi_elu1(x.w);
        zreg[0] += p[j][0]; zreg[1] += p[j][1];
        zreg[2] += p[j][2]; zreg[3] += p[j][3];
        uint2 q;
        q.x = pk2bf(p[j][0], p[j][1]);
        q.y = pk2bf(p[j][2], p[j][3]);
        *(uint2*)&sPhi[(r0 + j) * DKc + c0] = q;      // ds_store_b64
      }
      if (it + 1 < ITERS)
        __builtin_prefetch(Kbh + (srow0 + ST + r0) * DKc + c0, 0, 0);
#pragma unroll
      for (int i = 0; i < 4; ++i) {                   // transposed copy (GEMM2-A)
        uint2 q;
        q.x = pk2bf(p[0][i], p[1][i]);
        q.y = pk2bf(p[2][i], p[3][i]);
        *(uint2*)&sPhiT[(c0 + i) * ST + r0] = q;      // ds_store_b64
      }
    }
    __syncthreads();

    // ---- Phase 2: Vex = phi @ M (WMMA), E = V - Vex, store E^T ----
#pragma unroll
    for (int tp = 0; tp < 2; ++tp) {
      const int tile  = wave * 2 + tp;   // 16 output tiles (2 s x 8 v)
      const int stile = tile >> 3;
      const int vt    = tile & 7;
      v8f acc = (v8f){0.f, 0.f, 0.f, 0.f, 0.f, 0.f, 0.f, 0.f};
      const int arow = stile * 16 + n;
#pragma unroll
      for (int kt = 0; kt < 4; ++kt) {
        Frag a, b;
        const unsigned short* ap = &sPhi[arow * DKc + kt * 32 + hi * 8];
        a.q[0] = *(const uint4*)ap;             // K 0..7 (+hi*8)
        a.q[1] = *(const uint4*)(ap + 16);      // K 16..23 (+hi*8)
        const unsigned short* bp = &sMT[(vt * 16 + n) * DKc + kt * 32 + hi * 8];
        b.q[0] = *(const uint4*)bp;
        b.q[1] = *(const uint4*)(bp + 16);
        acc = __builtin_amdgcn_wmma_f32_16x16x32_bf16(
                  false, a.v, false, b.v, (short)0, acc, false, false);
      }
      // E = V - Vex: lane reads its (row, col) elements directly (64B/16 lanes)
      const float* vp = Vbh + (srow0 + stile * 16 + hi * 8) * DVc + vt * 16 + n;
      float e[8];
#pragma unroll
      for (int r = 0; r < 8; ++r) e[r] = vp[r * DVc] - acc[r];
      uint4 eq;
      eq.x = pk2bf(e[0], e[1]); eq.y = pk2bf(e[2], e[3]);
      eq.z = pk2bf(e[4], e[5]); eq.w = pk2bf(e[6], e[7]);
      *(uint4*)&sET[(vt * 16 + n) * ST + stile * 16 + hi * 8] = eq;
    }
    __syncthreads();

    // ---- Phase 3: delta += phi^T @ E (WMMA, per-wave accumulators) ----
    {
      Frag a;
      const unsigned short* ap = &sPhiT[(wave * 16 + n) * ST + hi * 8];
      a.q[0] = *(const uint4*)ap;
      a.q[1] = *(const uint4*)(ap + 16);
#pragma unroll
      for (int vt = 0; vt < 8; ++vt) {
        Frag b;
        const unsigned short* bp = &sET[(vt * 16 + n) * ST + hi * 8];
        b.q[0] = *(const uint4*)bp;
        b.q[1] = *(const uint4*)(bp + 16);
        accD[vt] = __builtin_amdgcn_wmma_f32_16x16x32_bf16(
                       false, a.v, false, b.v, (short)0, accD[vt], false, false);
      }
    }
    __syncthreads();
  }

  // ---- Epilogue: combine split-S partials ----
  // delta_M: C/D layout => element(vgpr r, lane) = (M = r + 8*hi, N = n)
#pragma unroll
  for (int vt = 0; vt < 8; ++vt) {
#pragma unroll
    for (int r = 0; r < 8; ++r) {
      const int k = wave * 16 + hi * 8 + r;
      const int v = vt * 16 + n;
      atomicAdd(&outM[((long)bh * DKc + k) * DVc + v], accD[vt][r]);
    }
  }
  // z: reduce per-thread partials through LDS, then one atomic per k
#pragma unroll
  for (int i = 0; i < 4; ++i) atomicAdd(&sZ[c0 + i], zreg[i]);
  __syncthreads();
  if (tid < DKc) atomicAdd(&outZ[bh * DKc + tid], sZ[tid]);
}

extern "C" void kernel_launch(void* const* d_in, const int* in_sizes, int n_in,
                              void* d_out, int out_size, void* d_ws, size_t ws_size,
                              hipStream_t stream) {
  (void)in_sizes; (void)n_in; (void)out_size; (void)d_ws; (void)ws_size;
  const float* K = (const float*)d_in[0];
  const float* V = (const float*)d_in[1];
  const float* M = (const float*)d_in[2];
  const float* z = (const float*)d_in[3];

  float* outM = (float*)d_out;
  float* outZ = outM + (size_t)BHc * DKc * DVc;

  // new_M = M + delta, new_z = z + colsum(phi): seed outputs with M and z,
  // then accumulate partials on top (d2d async copies are graph-capture safe).
  hipMemcpyAsync(outM, M, sizeof(float) * (size_t)BHc * DKc * DVc,
                 hipMemcpyDeviceToDevice, stream);
  hipMemcpyAsync(outZ, z, sizeof(float) * (size_t)BHc * DKc,
                 hipMemcpyDeviceToDevice, stream);

  delta_rule_kernel<<<BHc * SPLIT, 256, 0, stream>>>(K, V, M, outM, outZ);
}
